// SelfAttentionModule_4174708212457
// MI455X (gfx1250) — matrix operand order repaired and checked by reading, not compiled, not run on hardware
//
#include <hip/hip_runtime.h>
#include <hip/hip_bf16.h>
#include <math.h>

// ---------------------------------------------------------------------------
// Problem constants (from reference)
// ---------------------------------------------------------------------------
#define B_  4
#define S_  1024
#define D_  1024
#define H_  16
#define HD_ 64
// dist_emb has 2*1024-1 = 2047 rows; we pad to 2048 with a zero row so the
// 48-wide distance band never reads out of bounds.
#define EROWS 2048

typedef __attribute__((ext_vector_type(16))) __bf16 bf16x16;
typedef __attribute__((ext_vector_type(8)))  __bf16 bf16x8;
typedef __attribute__((ext_vector_type(8)))  float  f32x8;

// ---------------------------------------------------------------------------
// WMMA helper: D = A(16x32 bf16) * B(32x16 bf16) + C(16x16 f32)
// ---------------------------------------------------------------------------
static __device__ __forceinline__ f32x8 wmma_bf16(bf16x16 a, bf16x16 b, f32x8 c) {
  return __builtin_amdgcn_wmma_f32_16x16x32_bf16(
      /*neg_a=*/false, a, /*neg_b=*/false, b,
      /*c_mod=*/(short)0, c, /*reuse_a=*/false, /*reuse_b=*/false);
}

// A-operand (16x32, 16-bit): lane = row M (lane&15).
//   lanes 0-15 : elems 0-7 = K 0-7,  elems 8-15 = K 16-23
//   lanes 16-31: elems 0-7 = K 8-15, elems 8-15 = K 24-31
// Row data contiguous in K -> two 16B loads per lane.
static __device__ __forceinline__ bf16x16 load_A(const __bf16* __restrict__ base,
                                                 int stride, int lane) {
  const __bf16* p = base + (size_t)(lane & 15) * stride + ((lane >> 4) << 3);
  bf16x8 lo = *(const bf16x8*)(p);
  bf16x8 hi = *(const bf16x8*)(p + 16);
  return __builtin_shufflevector(lo, hi, 0, 1, 2, 3, 4, 5, 6, 7,
                                 8, 9, 10, 11, 12, 13, 14, 15);
}

// B-operand (32x16, 16-bit): lane = col N (lane&15).
//   lanes 0-15 : elems 0-7 = K 0-7,  elems 8-15 = K 16-23
//   lanes 16-31: elems 0-7 = K 8-15, elems 8-15 = K 24-31
static __device__ __forceinline__ bf16x16 load_B(const __bf16* __restrict__ base,
                                                 int stride, int lane) {
  const __bf16* p = base + (size_t)(lane & 15) * stride + ((lane >> 4) << 3);
  bf16x8 lo = *(const bf16x8*)(p);
  bf16x8 hi = *(const bf16x8*)(p + 16);
  return __builtin_shufflevector(lo, hi, 0, 1, 2, 3, 4, 5, 6, 7,
                                 8, 9, 10, 11, 12, 13, 14, 15);
}

// ---------------------------------------------------------------------------
// Conversion / layout kernels
// ---------------------------------------------------------------------------
__global__ void cvt_f32_bf16(const float* __restrict__ in, __bf16* __restrict__ out, int n) {
  int i = blockIdx.x * blockDim.x + threadIdx.x;
  if (i < n) out[i] = (__bf16)in[i];
}

// W [k,n] f32 -> WT [n,k] bf16 (so GEMM B-operand reads contiguous K runs).
__global__ void cvt_w_transpose(const float* __restrict__ wq, const float* __restrict__ wk,
                                const float* __restrict__ wv, __bf16* __restrict__ out) {
  int z = blockIdx.y;
  int t = blockIdx.x * blockDim.x + threadIdx.x;   // t = n*1024 + k
  if (t >= D_ * D_) return;
  const float* w = (z == 0) ? wq : (z == 1) ? wk : wv;
  int n = t >> 10, k = t & 1023;
  out[(size_t)z * D_ * D_ + t] = (__bf16)w[(size_t)k * D_ + n];
}

__global__ void cvt_emb(const float* __restrict__ de, __bf16* __restrict__ out) {
  int i = blockIdx.x * blockDim.x + threadIdx.x;   // 2048*64 elems
  if (i >= EROWS * HD_) return;
  int row = i >> 6;
  out[i] = (row < 2 * S_ - 1) ? (__bf16)de[i] : (__bf16)0.0f;
}

// ---------------------------------------------------------------------------
// Fused QKV projection GEMM.  One wave computes a 64(M) x 64(N) tile.
//   grid = (M/64=64, N/64=16, 3), block = 32 (one wave32)
//   z=0 -> Q [B,H,S,64] bf16; z=1 -> K [B,H,S,64] bf16; z=2 -> V^T [B,H,64,S]
// ---------------------------------------------------------------------------
__global__ __launch_bounds__(32) void qkv_gemm(
    const __bf16* __restrict__ Hb, const __bf16* __restrict__ WT,
    const float* __restrict__ bq, const float* __restrict__ bk,
    const float* __restrict__ bv,
    __bf16* __restrict__ Q, __bf16* __restrict__ K, __bf16* __restrict__ Vt) {
  const int lane = threadIdx.x;
  const int M0 = blockIdx.x * 64;
  const int N0 = blockIdx.y * 64;
  const int z  = blockIdx.z;
  const __bf16* W = WT + (size_t)z * D_ * D_;
  const float* bias = (z == 0) ? bq : (z == 1) ? bk : bv;

  f32x8 acc[4][4];
#pragma unroll
  for (int ms = 0; ms < 4; ++ms)
#pragma unroll
    for (int g = 0; g < 4; ++g) acc[ms][g] = (f32x8){};

  for (int kc = 0; kc < D_; kc += 32) {
    bf16x16 a[4], bm[4];
#pragma unroll
    for (int ms = 0; ms < 4; ++ms)
      a[ms] = load_A(Hb + (size_t)(M0 + ms * 16) * D_ + kc, D_, lane);
#pragma unroll
    for (int g = 0; g < 4; ++g)
      bm[g] = load_B(W + (size_t)(N0 + g * 16) * D_ + kc, D_, lane);
#pragma unroll
    for (int ms = 0; ms < 4; ++ms)
#pragma unroll
      for (int g = 0; g < 4; ++g)
        acc[ms][g] = wmma_bf16(a[ms], bm[g], acc[ms][g]);
  }

  const int colq = lane & 15, rb8 = (lane >> 4) << 3;
#pragma unroll
  for (int g = 0; g < 4; ++g) {
    int n = N0 + g * 16 + colq;
    float bs = bias[n];
    int h = n >> 6, d = n & 63;
#pragma unroll
    for (int ms = 0; ms < 4; ++ms)
#pragma unroll
      for (int v = 0; v < 8; ++v) {
        int m = M0 + ms * 16 + rb8 + v;
        int bb = m >> 10, s = m & 1023;
        float val = acc[ms][g][v] + bs;
        if (z == 0)
          Q[(((size_t)bb * H_ + h) * S_ + s) * HD_ + d] = (__bf16)val;
        else if (z == 1)
          K[(((size_t)bb * H_ + h) * S_ + s) * HD_ + d] = (__bf16)val;
        else
          Vt[(((size_t)bb * H_ + h) * HD_ + d) * S_ + s] = (__bf16)val;
      }
  }
}

// ---------------------------------------------------------------------------
// Fused flash-style attention with relative_key_query positional bias.
//   grid = B*H*(S/16) = 4096 blocks, block = 32 (one wave32).
//   Each wave owns 16 query rows; streams over r in steps of 32.
//   Streaming identity:  ctx = (sum_r e^{s_r} * prior_r * v_r) / sum_r e^{s_r}
//
// Cache policy: K/V/E (~16 MB, reused ~64x) must live in the 192 MB L2;
// `prior` (256 MB, single use) is loaded non-temporally and prefetched one
// r-step ahead so its HBM latency hides behind the 26 WMMAs of each step.
// ---------------------------------------------------------------------------
__global__ __launch_bounds__(32) void attn_kernel(
    const __bf16* __restrict__ Q, const __bf16* __restrict__ K,
    const __bf16* __restrict__ Vt, const __bf16* __restrict__ E,
    const float* __restrict__ amask, const float* __restrict__ prior,
    const float* __restrict__ hmaskp, float* __restrict__ out) {
  __shared__ float sQE[16 * 48];   // QE band tile  [l, t]
  __shared__ float sKE[32 * 48];   // KE band tile  [r, t]
  __shared__ float sP[16 * 32];    // probs*prior   [l, r] (transpose staging)

  const int lane = threadIdx.x;
  const int x = blockIdx.x;
  const int lt = x & 63, h = (x >> 6) & 15, b = x >> 10;
  const int L0 = lt << 4;
  const int bh = b * H_ + h;
  const int colq = lane & 15, rb8 = (lane >> 4) << 3;

  // Q tile as two A-operands (K = d 0-31, 32-63); constant over r loop.
  const __bf16* qb = Q + ((size_t)bh * S_ + L0) * HD_;
  const bf16x16 aq0 = load_A(qb, HD_, lane);
  const bf16x16 aq1 = load_A(qb + 32, HD_, lane);

  f32x8 acc0 = (f32x8){}, acc1 = (f32x8){}, acc2 = (f32x8){}, acc3 = (f32x8){};
  float m[8], l[8];
#pragma unroll
  for (int v = 0; v < 8; ++v) { m[v] = -__builtin_inff(); l[v] = 0.0f; }

  const float* mrow = amask + (size_t)b * S_;
  const float* prow = prior + ((size_t)bh * S_ + L0) * S_;

  // Warm the prefetch pipe for the first prior tile.
#pragma unroll
  for (int v = 0; v < 8; ++v)
    __builtin_prefetch(&prow[(size_t)(rb8 + v) * S_ + colq * 4], 0, 0);

  for (int R0 = 0; R0 < S_; R0 += 32) {
    // Prefetch next r-step's prior tile (the dominant HBM stream).
    if (R0 + 32 < S_) {
#pragma unroll
      for (int v = 0; v < 8; ++v)
        __builtin_prefetch(&prow[(size_t)(rb8 + v) * S_ + R0 + 32 + colq * 4], 0, 0);
    }

    const __bf16* kb = K + ((size_t)bh * S_ + R0) * HD_;

    // ---- raw q·k scores: two 16x16 tiles (r 0-15 and 16-31) ----
    f32x8 s0 = (f32x8){}, s1 = (f32x8){};
    s0 = wmma_bf16(aq0, load_B(kb, HD_, lane), s0);
    s0 = wmma_bf16(aq1, load_B(kb + 32, HD_, lane), s0);
    s1 = wmma_bf16(aq0, load_B(kb + 16 * HD_, HD_, lane), s1);
    s1 = wmma_bf16(aq1, load_B(kb + 16 * HD_ + 32, HD_, lane), s1);

    // ---- positional band: distances t = l-r+1023 span [tLo, tLo+46] ----
    const int tLo = L0 - R0 + 992;     // always in [0, 2000]; +47 <= 2047 (zero row)
    bf16x16 be[3][2];
#pragma unroll
    for (int ct = 0; ct < 3; ++ct) {
      be[ct][0] = load_B(E + (size_t)(tLo + ct * 16) * HD_, HD_, lane);
      be[ct][1] = load_B(E + (size_t)(tLo + ct * 16) * HD_ + 32, HD_, lane);
    }
    // QE = Q[16,64] @ E_band^T -> [16,48]
#pragma unroll
    for (int ct = 0; ct < 3; ++ct) {
      f32x8 qe = (f32x8){};
      qe = wmma_bf16(aq0, be[ct][0], qe);
      qe = wmma_bf16(aq1, be[ct][1], qe);
#pragma unroll
      for (int v = 0; v < 8; ++v) sQE[(rb8 + v) * 48 + ct * 16 + colq] = qe[v];
    }
    // KE = K[32,64] @ E_band^T -> [32,48]
#pragma unroll
    for (int rs = 0; rs < 2; ++rs) {
      bf16x16 ak0 = load_A(kb + rs * 16 * HD_, HD_, lane);
      bf16x16 ak1 = load_A(kb + rs * 16 * HD_ + 32, HD_, lane);
#pragma unroll
      for (int ct = 0; ct < 3; ++ct) {
        f32x8 ke = (f32x8){};
        ke = wmma_bf16(ak0, be[ct][0], ke);
        ke = wmma_bf16(ak1, be[ct][1], ke);
#pragma unroll
        for (int v = 0; v < 8; ++v)
          sKE[(rs * 16 + rb8 + v) * 48 + ct * 16 + colq] = ke[v];
      }
    }
    __syncthreads();   // single-wave WG: S_NOP, LDS is in-order per wave

    // ---- assemble scores (diagonal gather), scale, mask, online softmax ----
    const float am0 = mrow[R0 + colq];
    const float am1 = mrow[R0 + 16 + colq];
    float e0[8], e1[8];
#pragma unroll
    for (int v = 0; v < 8; ++v) {
      const int li = rb8 + v;
      const int ri0 = colq, ri1 = 16 + colq;
      const int i0 = li - ri0 + 31;   // band column index, in [16,46]
      const int i1 = li - ri1 + 31;   // in [0,30]
      float t0 = (s0[v] + sQE[li * 48 + i0] + sKE[ri0 * 48 + i0]) * 0.125f + am0;
      float t1 = (s1[v] + sQE[li * 48 + i1] + sKE[ri1 * 48 + i1]) * 0.125f + am1;
      // row max over the 16 lanes holding this row (butterfly stays in half)
      float rm = fmaxf(t0, t1);
      rm = fmaxf(rm, __shfl_xor(rm, 1));
      rm = fmaxf(rm, __shfl_xor(rm, 2));
      rm = fmaxf(rm, __shfl_xor(rm, 4));
      rm = fmaxf(rm, __shfl_xor(rm, 8));
      const float mn = fmaxf(m[v], rm);
      const float sc = __expf(m[v] - mn);
      const float ee0 = __expf(t0 - mn);
      const float ee1 = __expf(t1 - mn);
      float rs_ = ee0 + ee1;
      rs_ += __shfl_xor(rs_, 1);
      rs_ += __shfl_xor(rs_, 2);
      rs_ += __shfl_xor(rs_, 4);
      rs_ += __shfl_xor(rs_, 8);
      l[v] = l[v] * sc + rs_;
      m[v] = mn;
      e0[v] = ee0; e1[v] = ee1;
      acc0[v] *= sc; acc1[v] *= sc; acc2[v] *= sc; acc3[v] *= sc;
    }

    // ---- p = exp * prior -> LDS, then transpose-read into A layout ----
    // prior is a 256 MB single-use stream: non-temporal loads keep it from
    // evicting the K/V/E working set out of L2.
#pragma unroll
    for (int v = 0; v < 8; ++v) {
      const int li = rb8 + v;
      const float pr0 = __builtin_nontemporal_load(&prow[(size_t)li * S_ + R0 + colq]);
      const float pr1 = __builtin_nontemporal_load(&prow[(size_t)li * S_ + R0 + 16 + colq]);
      sP[li * 32 + colq]      = e0[v] * pr0;
      sP[li * 32 + 16 + colq] = e1[v] * pr1;
    }
    __syncthreads();

    bf16x16 ap;
    const int koff = (lane >> 4) << 3;
#pragma unroll
    for (int i = 0; i < 8; ++i) {
      ap[i]     = (__bf16)sP[colq * 32 + koff + i];
      ap[i + 8] = (__bf16)sP[colq * 32 + 16 + koff + i];
    }

    // ---- acc += P[16,32] @ V[32,64]  (V^T layout gives contiguous B reads) ----
    const __bf16* vb = Vt + (size_t)bh * HD_ * S_ + R0;
    acc0 = wmma_bf16(ap, load_B(vb + 0 * 16 * S_, S_, lane), acc0);
    acc1 = wmma_bf16(ap, load_B(vb + 1 * 16 * S_, S_, lane), acc1);
    acc2 = wmma_bf16(ap, load_B(vb + 2 * 16 * S_, S_, lane), acc2);
    acc3 = wmma_bf16(ap, load_B(vb + 3 * 16 * S_, S_, lane), acc3);
    __syncthreads();
  }

  // ---- epilogue: normalize, apply head_mask, write [B,S,H*64] fp32 ----
  const float hm = hmaskp[h];
  float* op = out + ((size_t)b * S_ + L0) * D_ + h * HD_;
#pragma unroll
  for (int v = 0; v < 8; ++v) {
    const int li = rb8 + v;
    const float inv = hm / l[v];
    __builtin_nontemporal_store(acc0[v] * inv, &op[(size_t)li * D_ + 0 + colq]);
    __builtin_nontemporal_store(acc1[v] * inv, &op[(size_t)li * D_ + 16 + colq]);
    __builtin_nontemporal_store(acc2[v] * inv, &op[(size_t)li * D_ + 32 + colq]);
    __builtin_nontemporal_store(acc3[v] * inv, &op[(size_t)li * D_ + 48 + colq]);
  }
}

// ---------------------------------------------------------------------------
// Host-side launcher
// ---------------------------------------------------------------------------
extern "C" void kernel_launch(void* const* d_in, const int* in_sizes, int n_in,
                              void* d_out, int out_size, void* d_ws, size_t ws_size,
                              hipStream_t stream) {
  const float* hs    = (const float*)d_in[0];
  const float* amask = (const float*)d_in[1];
  const float* prior = (const float*)d_in[2];
  const float* hmask = (const float*)d_in[3];
  const float* Wq    = (const float*)d_in[4];
  const float* bq    = (const float*)d_in[5];
  const float* Wk    = (const float*)d_in[6];
  const float* bk    = (const float*)d_in[7];
  const float* Wv    = (const float*)d_in[8];
  const float* bv    = (const float*)d_in[9];
  const float* de    = (const float*)d_in[10];
  float* out = (float*)d_out;

  // Workspace carve-up (~40.1 MB total, 256B aligned)
  char* ws = (char*)d_ws;
  size_t off = 0;
  auto carve = [&](size_t bytes) {
    char* p = ws + off;
    off = (off + bytes + 255) & ~(size_t)255;
    return p;
  };
  __bf16* Hb = (__bf16*)carve((size_t)B_ * S_ * D_ * 2);        // 8 MB
  __bf16* WT = (__bf16*)carve((size_t)3 * D_ * D_ * 2);         // 6 MB
  __bf16* Eb = (__bf16*)carve((size_t)EROWS * HD_ * 2);         // 256 KB
  __bf16* Qb = (__bf16*)carve((size_t)B_ * H_ * S_ * HD_ * 2);  // 8 MB
  __bf16* Kb = (__bf16*)carve((size_t)B_ * H_ * S_ * HD_ * 2);  // 8 MB
  __bf16* Vt = (__bf16*)carve((size_t)B_ * H_ * HD_ * S_ * 2);  // 8 MB

  // 1) conversions
  {
    int n = B_ * S_ * D_;
    cvt_f32_bf16<<<(n + 255) / 256, 256, 0, stream>>>(hs, Hb, n);
  }
  {
    dim3 g((D_ * D_ + 255) / 256, 3);
    cvt_w_transpose<<<g, 256, 0, stream>>>(Wq, Wk, Wv, WT);
  }
  {
    int n = EROWS * HD_;
    cvt_emb<<<(n + 255) / 256, 256, 0, stream>>>(de, Eb);
  }

  // 2) QKV projection GEMMs (M=4096, N=1024, K=1024, x3)
  {
    dim3 g(64, 16, 3);
    qkv_gemm<<<g, 32, 0, stream>>>(Hb, WT, bq, bk, bv, Qb, Kb, Vt);
  }

  // 3) fused attention (B*H*(S/16) waves)
  {
    dim3 g(B_ * H_ * (S_ / 16));
    attn_kernel<<<g, 32, 0, stream>>>(Qb, Kb, Vt, Eb, amask, prior, hmask, out);
  }
}
